// OccPerformerEncoder_v2_69080253989536
// MI455X (gfx1250) — compile-verified, matrix operand order checked by reading, not run on hardware
//
#include <hip/hip_runtime.h>
#include <math.h>

// ---------------- constants ----------------
#define NTOK 2049
#define NPAD 2080                 // multiple of 32 (WMMA K), 16 (tiles)
#define MPAD (8*NPAD)             // 16640 flattened rows
#define TMM  (MPAD/16)            // 1040 row tiles
#define EMB  256
#define NH   8
#define DKH  32
#define RFFD 128
#define F2D  256                  // 2*RFF per head
#define GN_G 8

typedef unsigned short u16b;      // bf16 bits in memory
typedef __attribute__((ext_vector_type(16))) __bf16 v16bf;
typedef __attribute__((ext_vector_type(8)))  __bf16 v8bf;
typedef __attribute__((ext_vector_type(8)))  float  v8f;

// ---------------- device helpers ----------------
__device__ __forceinline__ u16b f2bf(float x) {
    union { float f; unsigned u; } v; v.f = x;
    unsigned u = v.u;
    u += 0x7FFFu + ((u >> 16) & 1u);
    return (u16b)(u >> 16);
}
__device__ __forceinline__ float bf2f(u16b b) {
    union { unsigned u; float f; } v; v.u = ((unsigned)b) << 16;
    return v.f;
}
__device__ __forceinline__ float gelu_f(float x) {
    return 0.5f * x * (1.0f + erff(x * 0.70710678118654752f));
}

// A-fragment (and symmetric B-fragment) loader for v_wmma_f32_16x16x32_bf16.
// lane L: row = L%16, k-chunk base kb = (L/16)*8; elements 0..7 = K kb..kb+7,
// elements 8..15 = K kb+16..kb+23.
__device__ __forceinline__ v16bf load_frag(const u16b* __restrict__ base, int stride, int lane) {
    const u16b* p = base + (size_t)(lane & 15) * stride + ((lane >> 4) << 3);
    v8bf lo = *(const v8bf*)(p);
    v8bf hi = *(const v8bf*)(p + 16);
    v16bf f;
#pragma unroll
    for (int i = 0; i < 8; ++i) { f[i] = lo[i]; f[i + 8] = hi[i]; }
    return f;
}

__device__ __forceinline__ v8f wmma_tile(const u16b* __restrict__ A, int lda,
                                         const u16b* __restrict__ Bt, int ldb,
                                         int K, int lane) {
    v8f acc = {0.f,0.f,0.f,0.f,0.f,0.f,0.f,0.f};
    for (int k = 0; k < K; k += 32) {
        v16bf a = load_frag(A + k, lda, lane);
        v16bf b = load_frag(Bt + k, ldb, lane);
        acc = __builtin_amdgcn_wmma_f32_16x16x32_bf16(false, a, false, b, (short)0, acc, false, false);
    }
    return acc;
}

// ---------------- generic GEMM: C = A(MxK) * Bt(NxK)^T [+bias][+resid][gelu] ----------------
template<int ACT, int RESID>
__global__ void gemm_kernel(const u16b* __restrict__ A, int lda, long sAz,
                            const u16b* __restrict__ Bt, int ldb, long sBz,
                            float* __restrict__ Cf, u16b* __restrict__ Cb, int ldc, long sCz,
                            const float* __restrict__ bias,
                            const float* __restrict__ resid,
                            int tilesM, int tilesN, int K) {
    int lane = threadIdx.x & 31;
    int gw = blockIdx.x * 8 + (threadIdx.x >> 5);
    if (gw >= tilesM * tilesN) return;
    int tm = gw / tilesN, tn = gw % tilesN;
    long z = blockIdx.z;
    const u16b* Ap = A + z * sAz + (size_t)tm * 16 * lda;
    const u16b* Bp = Bt + z * sBz + (size_t)tn * 16 * ldb;
    v8f acc = wmma_tile(Ap, lda, Bp, ldb, K, lane);
    int col = tn * 16 + (lane & 15);
    int mb = tm * 16 + ((lane >> 4) << 3);
    float bv = bias ? bias[col] : 0.f;
#pragma unroll
    for (int r = 0; r < 8; ++r) {
        long m = mb + r;
        float v = acc[r] + bv;
        long o = z * sCz + m * (long)ldc + col;
        if (RESID) v += resid[o];
        if (ACT == 1) v = gelu_f(v);
        if (Cf) Cf[o] = v;
        if (Cb) Cb[o] = f2bf(v);
    }
}

// ---------------- patch-embed GEMM: writes into xseq with PE ----------------
__global__ void patch_gemm_kernel(const u16b* __restrict__ A, const u16b* __restrict__ Bt,
                                  const float* __restrict__ bias, const float* __restrict__ pe,
                                  float* __restrict__ xseq, int K, int base) {
    const int tilesN = EMB / 16;     // 16
    const int tilesM = 512;          // M = 8192
    int lane = threadIdx.x & 31;
    int gw = blockIdx.x * 8 + (threadIdx.x >> 5);
    if (gw >= tilesM * tilesN) return;
    int tm = gw / tilesN, tn = gw % tilesN;
    v8f acc = wmma_tile(A + (size_t)tm * 16 * K, K, Bt + (size_t)tn * 16 * K, K, K, lane);
    int col = tn * 16 + (lane & 15);
    int mb = tm * 16 + ((lane >> 4) << 3);
#pragma unroll
    for (int r = 0; r < 8; ++r) {
        int m = mb + r;
        int b = m >> 10, t = m & 1023;
        long row = (long)b * NPAD + base + t;
        xseq[row * EMB + col] = acc[r] + bias[col] + pe[t * EMB + col];
    }
}

// ---------------- phi RFF GEMM (per head), cos/sin epilogue ----------------
template<int TRANS>
__global__ void phi_gemm_kernel(const u16b* __restrict__ qr, const u16b* __restrict__ WrT,
                                const float* __restrict__ br, u16b* __restrict__ out) {
    const int tilesN = RFFD / 16;    // 8
    int lane = threadIdx.x & 31;
    int h = blockIdx.z;
    int gw = blockIdx.x * 8 + (threadIdx.x >> 5);
    if (gw >= TMM * tilesN) return;
    int tm = gw / tilesN, tn = gw % tilesN;
    v8f acc = wmma_tile(qr + (size_t)tm * 16 * EMB + h * DKH, EMB,
                        WrT + (size_t)h * RFFD * DKH + (size_t)tn * 16 * DKH, DKH, DKH, lane);
    const float inv = 0.08838834764831845f;   // 1/sqrt(128)
    int f = tn * 16 + (lane & 15);
    int mb = tm * 16 + ((lane >> 4) << 3);
#pragma unroll
    for (int r = 0; r < 8; ++r) {
        int m = mb + r;
        int b = m / NPAD, n = m % NPAD;
        float p = acc[r] + br[h * RFFD + f];
        float c = cosf(p) * inv, s = sinf(p) * inv;
        if (n >= NTOK) { c = 0.f; s = 0.f; }
        if (!TRANS) {
            size_t o = ((size_t)m * NH + h) * F2D + f;
            out[o] = f2bf(c); out[o + RFFD] = f2bf(s);
        } else {
            size_t o = (((size_t)(b * NH + h) * F2D) + f) * NPAD + n;
            out[o] = f2bf(c); out[o + (size_t)RFFD * NPAD] = f2bf(s);
        }
    }
}

// ---------------- attention output GEMM per (b,h): (phi_q @ Kv) * z ----------------
__global__ void attn_gemm_kernel(const u16b* __restrict__ phiq, const u16b* __restrict__ Kvt,
                                 const float* __restrict__ denom, u16b* __restrict__ attnb) {
    const int tilesM = NPAD / 16;    // 130
    const int tilesN = DKH / 16;     // 2
    int lane = threadIdx.x & 31;
    int z = blockIdx.z; int b = z >> 3, h = z & 7;
    int gw = blockIdx.x * 8 + (threadIdx.x >> 5);
    if (gw >= tilesM * tilesN) return;
    int tm = gw / tilesN, tn = gw % tilesN;
    const u16b* A = phiq + (((size_t)b * NPAD) * NH + h) * F2D;
    const u16b* B = Kvt + (size_t)z * DKH * F2D;
    v8f acc = wmma_tile(A + (size_t)tm * 16 * (NH * F2D), NH * F2D,
                        B + (size_t)tn * 16 * F2D, F2D, F2D, lane);
    int col = tn * 16 + (lane & 15);
    int nb = tm * 16 + ((lane >> 4) << 3);
#pragma unroll
    for (int r = 0; r < 8; ++r) {
        int n = nb + r;
        float zz = 1.0f / (denom[((size_t)b * NPAD + n) * NH + h] + 1e-6f);
        attnb[((size_t)b * NPAD + n) * EMB + h * DKH + col] = f2bf(acc[r] * zz);
    }
}

// ---------------- layer norm (row per block, 256 cols) ----------------
// MODE 0: write bf16 (zero padded rows). MODE 1: write final output mapping.
template<int MODE>
__global__ void layernorm_kernel(const float* __restrict__ x, const float* __restrict__ sc,
                                 const float* __restrict__ bi, u16b* __restrict__ outb,
                                 float* __restrict__ outf) {
    __shared__ float red[256];
    int row = blockIdx.x, tid = threadIdx.x;
    float v = x[(size_t)row * EMB + tid];
    red[tid] = v; __syncthreads();
    for (int s = 128; s > 0; s >>= 1) { if (tid < s) red[tid] += red[tid + s]; __syncthreads(); }
    float mean = red[0] / (float)EMB; __syncthreads();
    float d = v - mean;
    red[tid] = d * d; __syncthreads();
    for (int s = 128; s > 0; s >>= 1) { if (tid < s) red[tid] += red[tid + s]; __syncthreads(); }
    float rstd = rsqrtf(red[0] / (float)EMB + 1e-5f);
    float y = d * rstd * sc[tid] + bi[tid];
    int b = row / NPAD, n = row % NPAD;
    if (MODE == 0) {
        outb[(size_t)row * EMB + tid] = (n < NTOK) ? f2bf(y) : (u16b)0;
    } else {
        if (n == 0) outf[(size_t)b * EMB + tid] = y;
        else if (n < NTOK) outf[8 * EMB + ((size_t)b * (NTOK - 1) + (n - 1)) * EMB + tid] = y;
    }
}

// ---------------- small elementwise / cache kernels ----------------
__global__ void affine_kernel(const float* __restrict__ g, const float* __restrict__ cs,
                              const float* __restrict__ cb, float* __restrict__ out) {
    long total = 8L * 4 * 512 * 512;
    long idx = (long)blockIdx.x * 256 + threadIdx.x;
    if (idx >= total) return;
    int c = (int)((idx / (512 * 512)) % 4);
    out[idx] = g[idx] * cs[c] + cb[c];
}

__global__ void ropec_kernel(float* __restrict__ rc) {
    int idx = blockIdx.x * 256 + threadIdx.x;   // NPAD*32
    if (idx >= NPAD * 32) return;
    int n = idx >> 5, i = idx & 31, j = i & 15;
    float fr = expf(-(float)j * (logf(10000.f) / 15.f));
    float a = (float)n * fr;
    rc[idx] = (i < 16) ? sinf(a) : cosf(a);
}

__global__ void rope_kernel(const float* __restrict__ q, const float* __restrict__ rc,
                            u16b* __restrict__ out) {
    long total = (long)MPAD * EMB;
    long idx = (long)blockIdx.x * 256 + threadIdx.x;
    if (idx >= total) return;
    int c = (int)(idx & 255); long m = idx >> 8;
    int h = c >> 5, j = c & 31;
    int n = (int)(m % NPAD);
    const float* qp = q + m * EMB + h * DKH;
    float r;
    if (j < 16) {
        float t1 = qp[j], t2 = qp[16 + j];
        r = t1 * rc[n * 32 + 16 + j] - t2 * rc[n * 32 + j];
    } else {
        int j2 = j - 16;
        float t1 = qp[j2], t2 = qp[16 + j2];
        r = t1 * rc[n * 32 + j2] + t2 * rc[n * 32 + 16 + j2];
    }
    out[idx] = f2bf(r);
}

__global__ void vt_kernel(const float* __restrict__ v, u16b* __restrict__ Vt) {
    long total = 8L * NH * DKH * NPAD;
    long idx = (long)blockIdx.x * 256 + threadIdx.x;
    if (idx >= total) return;
    int n = (int)(idx % NPAD); long t = idx / NPAD;
    int d = (int)(t % DKH); t /= DKH;
    int h = (int)(t % NH); int b = (int)(t / NH);
    int nn = (n < NTOK) ? 1 : 0;
    float val = nn ? v[((size_t)b * NPAD + n) * EMB + h * DKH + d] : 0.f;
    Vt[idx] = f2bf(val);
}

__global__ void ksum_kernel(const u16b* __restrict__ phiKt, float* __restrict__ ksum) {
    int idx = blockIdx.x * 256 + threadIdx.x;   // 8*NH*F2D = 16384
    if (idx >= 8 * NH * F2D) return;
    const u16b* p = phiKt + (size_t)idx * NPAD;
    float s = 0.f;
    for (int n = 0; n < NPAD; ++n) s += bf2f(p[n]);
    ksum[idx] = s;
}

__global__ void denom_kernel(const u16b* __restrict__ phiq, const float* __restrict__ ksum,
                             float* __restrict__ denom) {
    long total = 8L * NPAD * NH;
    long idx = (long)blockIdx.x * 256 + threadIdx.x;
    if (idx >= total) return;
    int h = (int)(idx % NH); long t = idx / NH;
    int n = (int)(t % NPAD); int b = (int)(t / NPAD);
    if (n >= NTOK) { denom[idx] = 0.f; return; }
    const u16b* p = phiq + (((size_t)b * NPAD + n) * NH + h) * F2D;
    const float* ks = ksum + ((size_t)b * NH + h) * F2D;
    float s = 0.f;
    for (int f = 0; f < F2D; ++f) s += bf2f(p[f]) * ks[f];
    denom[idx] = s;
}

__global__ void transpose_cvt_kernel(const float* __restrict__ src, u16b* __restrict__ dst,
                                     int R, int C, long sS, long dS) {
    long total = (long)R * C;
    long idx = (long)blockIdx.x * 256 + threadIdx.x;
    if (idx >= total) return;
    int c = (int)(idx % C); int r = (int)(idx / C);
    dst[(long)blockIdx.z * dS + (size_t)c * R + r] = f2bf(src[(long)blockIdx.z * sS + idx]);
}

__global__ void pe2d_kernel(float* __restrict__ pe) {
    int idx = blockIdx.x * 256 + threadIdx.x;   // 1024*256
    if (idx >= 1024 * 256) return;
    int col = idx & 255; int t = idx >> 8;
    int i = t >> 5, j = t & 31;
    int quad = col >> 6, qi = col & 63;
    float dv = expf(-(float)qi * (logf(10000.f) / 63.f));
    float pos = (quad < 2) ? (float)i : (float)j;
    float a = pos * dv;
    pe[idx] = ((quad & 1) == 0) ? sinf(a) : cosf(a);
}

__global__ void tcls_kernel(const int* __restrict__ ts, const float* __restrict__ tfw,
                            const float* __restrict__ tfb, const float* __restrict__ cls,
                            float* __restrict__ xseq) {
    __shared__ float tf[256];
    int b = blockIdx.x, tid = threadIdx.x;
    if (tid < 128) {
        float fr = expf(-(float)tid * (logf(10000.f) / 127.f));
        float a = (float)ts[b] * fr;
        tf[tid] = sinf(a); tf[tid + 128] = cosf(a);
    }
    __syncthreads();
    float g = tfb[tid], bet = tfb[tid + 256];
    for (int r = 0; r < 256; ++r) {
        float t = tf[r];
        g += t * tfw[r * 512 + tid];
        bet += t * tfw[r * 512 + 256 + tid];
    }
    xseq[((size_t)b * NPAD) * EMB + tid] = (1.f + g) * cls[tid] + bet;
}

__global__ void zeropad_kernel(float* __restrict__ xseq) {
    const int per = (NPAD - NTOK) * EMB;
    int idx = blockIdx.x * 256 + threadIdx.x;
    if (idx >= 8 * per) return;
    int b = idx / per; int r = idx % per;
    int n = NTOK + r / EMB; int c = r % EMB;
    xseq[((size_t)b * NPAD + n) * EMB + c] = 0.f;
}

__global__ void patchify_kernel(const float* __restrict__ x, u16b* __restrict__ pm,
                                int Himg, int P) {
    int cols = 128 * P * P;
    long total = 8192L * cols;
    long idx = (long)blockIdx.x * 256 + threadIdx.x;
    if (idx >= total) return;
    int col = (int)(idx % cols); long m = idx / cols;
    int b = (int)(m >> 10); int t = (int)(m & 1023);
    int i = t >> 5, j = t & 31;
    int pj = col % P; int t2 = col / P; int pi = t2 % P; int c = t2 / P;
    pm[idx] = f2bf(x[(((size_t)b * 128 + c) * Himg + i * P + pi) * Himg + j * P + pj]);
}

// ---------------- stem kernels ----------------
__global__ void conv3x3_s2_kernel(const float* __restrict__ in, const float* __restrict__ w,
                                  float* __restrict__ out, int Bn, int Cin, int Cout,
                                  int Hin, int Win) {
    int Ho = Hin >> 1, Wo = Win >> 1;
    long total = (long)Bn * Cout * Ho * Wo;
    long idx = (long)blockIdx.x * 256 + threadIdx.x;
    if (idx >= total) return;
    int wo = (int)(idx % Wo); long t = idx / Wo;
    int ho = (int)(t % Ho); t /= Ho;
    int co = (int)(t % Cout); int b = (int)(t / Cout);
    float acc = 0.f;
    for (int ci = 0; ci < Cin; ++ci) {
        const float* ip = in + (((size_t)b * Cin + ci) * Hin) * Win;
        const float* wp = w + ((size_t)co * Cin + ci) * 9;
#pragma unroll
        for (int kh = 0; kh < 3; ++kh) {
            int hi = ho * 2 + kh - 1;
            if ((unsigned)hi >= (unsigned)Hin) continue;
#pragma unroll
            for (int kw = 0; kw < 3; ++kw) {
                int wi = wo * 2 + kw - 1;
                if ((unsigned)wi >= (unsigned)Win) continue;
                acc += ip[(size_t)hi * Win + wi] * wp[kh * 3 + kw];
            }
        }
    }
    out[idx] = acc;
}

__global__ void dwconv3x3_kernel(const float* __restrict__ in, const float* __restrict__ w,
                                 float* __restrict__ out, int Bn, int C, int H, int W) {
    long total = (long)Bn * C * H * W;
    long idx = (long)blockIdx.x * 256 + threadIdx.x;
    if (idx >= total) return;
    int x = (int)(idx % W); long t = idx / W;
    int y = (int)(t % H); t /= H;
    int c = (int)(t % C); int b = (int)(t / C);
    const float* ip = in + ((size_t)b * C + c) * H * W;
    const float* wp = w + (size_t)c * 9;
    float acc = 0.f;
#pragma unroll
    for (int kh = 0; kh < 3; ++kh) {
        int hi = y + kh - 1;
        if ((unsigned)hi >= (unsigned)H) continue;
#pragma unroll
        for (int kw = 0; kw < 3; ++kw) {
            int wi = x + kw - 1;
            if ((unsigned)wi >= (unsigned)W) continue;
            acc += ip[(size_t)hi * W + wi] * wp[kh * 3 + kw];
        }
    }
    out[idx] = acc;
}

__global__ void gn_stats_kernel(const float* __restrict__ x, float* __restrict__ mean,
                                float* __restrict__ rstd, int C, int S) {
    __shared__ float r1[256], r2[256];
    int bg = blockIdx.x;
    int g = bg % GN_G; int b = bg / GN_G;
    int Cg = C / GN_G;
    long len = (long)Cg * S;
    const float* p = x + ((size_t)b * C + (size_t)g * Cg) * S;
    float s = 0.f, sq = 0.f;
    for (long i = threadIdx.x; i < len; i += 256) { float v = p[i]; s += v; sq += v * v; }
    r1[threadIdx.x] = s; r2[threadIdx.x] = sq; __syncthreads();
    for (int st = 128; st > 0; st >>= 1) {
        if ((int)threadIdx.x < st) { r1[threadIdx.x] += r1[threadIdx.x + st]; r2[threadIdx.x] += r2[threadIdx.x + st]; }
        __syncthreads();
    }
    if (threadIdx.x == 0) {
        float m = r1[0] / (float)len;
        float var = r2[0] / (float)len - m * m;
        mean[bg] = m; rstd[bg] = rsqrtf(var + 1e-5f);
    }
}

__global__ void gn_gelu_kernel(float* __restrict__ x, const float* __restrict__ mean,
                               const float* __restrict__ rstd, const float* __restrict__ gs,
                               const float* __restrict__ gb, int C, int S) {
    long total = 8L * C * S;
    long idx = (long)blockIdx.x * 256 + threadIdx.x;
    if (idx >= total) return;
    int c = (int)((idx / S) % C);
    int b = (int)(idx / ((long)C * S));
    int Cg = C / GN_G;
    int bg = b * GN_G + c / Cg;
    float y = (x[idx] - mean[bg]) * rstd[bg] * gs[c] + gb[c];
    x[idx] = gelu_f(y);
}

// ---------------- host ----------------
extern "C" void kernel_launch(void* const* d_in, const int* in_sizes, int n_in,
                              void* d_out, int out_size, void* d_ws, size_t ws_size,
                              hipStream_t stream) {
    (void)in_sizes; (void)n_in; (void)out_size; (void)ws_size;
    const float* grid      = (const float*)d_in[0];
    const int*   timesteps = (const int*)d_in[1];
    const float* chan_s    = (const float*)d_in[2];
    const float* chan_b    = (const float*)d_in[3];
    const float *cw[3], *g1s[3], *g1b[3], *dww[3], *g2s[3], *g2b[3];
    for (int s = 0; s < 3; ++s) {
        int base = 4 + s * 6;
        cw[s]  = (const float*)d_in[base + 0];
        g1s[s] = (const float*)d_in[base + 1];
        g1b[s] = (const float*)d_in[base + 2];
        dww[s] = (const float*)d_in[base + 3];
        g2s[s] = (const float*)d_in[base + 4];
        g2b[s] = (const float*)d_in[base + 5];
    }
    const float* p32w = (const float*)d_in[22]; const float* p32b = (const float*)d_in[23];
    const float* p16w = (const float*)d_in[24]; const float* p16b = (const float*)d_in[25];
    const float* clsw = (const float*)d_in[26];
    const float* tfw  = (const float*)d_in[27]; const float* tfb = (const float*)d_in[28];
    const float *n1s[6], *n1b[6], *wq[6], *wk[6], *wv[6], *pw[6], *pb[6], *Wr[6], *br[6];
    const float *n2s[6], *n2b[6], *m1w[6], *m1b[6], *m2w[6], *m2b[6];
    for (int l = 0; l < 6; ++l) {
        int base = 29 + l * 15;
        n1s[l] = (const float*)d_in[base + 0];  n1b[l] = (const float*)d_in[base + 1];
        wq[l]  = (const float*)d_in[base + 2];  wk[l]  = (const float*)d_in[base + 3];
        wv[l]  = (const float*)d_in[base + 4];  pw[l]  = (const float*)d_in[base + 5];
        pb[l]  = (const float*)d_in[base + 6];  Wr[l]  = (const float*)d_in[base + 7];
        br[l]  = (const float*)d_in[base + 8];  n2s[l] = (const float*)d_in[base + 9];
        n2b[l] = (const float*)d_in[base + 10]; m1w[l] = (const float*)d_in[base + 11];
        m1b[l] = (const float*)d_in[base + 12]; m2w[l] = (const float*)d_in[base + 13];
        m2b[l] = (const float*)d_in[base + 14];
    }
    const float* post_s = (const float*)d_in[119];
    const float* post_b = (const float*)d_in[120];

    char* ws = (char*)d_ws;
    // --- stem arena (explicit reuse) ---
    const size_t S_T0 = 8UL * 4 * 512 * 512 * 4;
    const size_t S_C1 = 8UL * 128 * 256 * 256 * 4;
    const size_t S_C2 = 8UL * 128 * 128 * 128 * 4;
    const size_t S_C3 = 8UL * 128 * 64 * 64 * 4;
    size_t o_t0  = 0;
    size_t o_c1a = o_t0 + S_T0;
    size_t o_c1b = o_c1a + S_C1;
    size_t o_c2a = o_c1a;                 // reuse c1a after dw1
    size_t o_c2b = o_c2a + S_C2;
    size_t o_c3a = o_c2b + S_C2;
    size_t o_c3b = o_c3a + S_C3;
    size_t o_pm32 = o_c3b + S_C3;
    size_t o_pm16 = o_pm32 + 8192UL * 2048 * 2;
    size_t o_xseq = o_pm16 + 8192UL * 512 * 2;
    size_t o_scr  = o_xseq + (size_t)MPAD * EMB * 4;
    float* t0  = (float*)(ws + o_t0);
    float* c1a = (float*)(ws + o_c1a);
    float* c1b = (float*)(ws + o_c1b);
    float* c2a = (float*)(ws + o_c2a);
    float* c2b = (float*)(ws + o_c2b);
    float* c3a = (float*)(ws + o_c3a);
    float* c3b = (float*)(ws + o_c3b);
    u16b*  pm32 = (u16b*)(ws + o_pm32);
    u16b*  pm16 = (u16b*)(ws + o_pm16);
    float* xseq = (float*)(ws + o_xseq);

    // --- transformer scratch (used only after stem; may overlap c1b region in time) ---
    size_t cur = (o_scr + 255) & ~(size_t)255;
    auto BUMP = [&](size_t bytes) -> char* {
        char* p = ws + cur; cur = (cur + bytes + 255) & ~(size_t)255; return p;
    };
    u16b*  xlnb  = (u16b*) BUMP((size_t)MPAD * EMB * 2);
    float* qf    = (float*)BUMP((size_t)MPAD * EMB * 4);
    float* kf    = (float*)BUMP((size_t)MPAD * EMB * 4);
    float* vf    = (float*)BUMP((size_t)MPAD * EMB * 4);
    u16b*  qrb   = (u16b*) BUMP((size_t)MPAD * EMB * 2);
    u16b*  krb   = (u16b*) BUMP((size_t)MPAD * EMB * 2);
    u16b*  phiq  = (u16b*) BUMP((size_t)MPAD * NH * F2D * 2);
    u16b*  phiKt = (u16b*) BUMP(64UL * F2D * NPAD * 2);
    u16b*  Vt    = (u16b*) BUMP(64UL * DKH * NPAD * 2);
    float* ksum  = (float*)BUMP(64UL * F2D * 4);
    float* Kv    = (float*)BUMP(64UL * F2D * DKH * 4);
    u16b*  Kvt   = (u16b*) BUMP(64UL * F2D * DKH * 2);
    float* denom = (float*)BUMP((size_t)MPAD * NH * 4);
    u16b*  attnb = (u16b*) BUMP((size_t)MPAD * EMB * 2);
    u16b*  h1    = (u16b*) BUMP((size_t)MPAD * 1024 * 2);

    // --- persistent region above stem peak ---
    size_t pcur = ((o_c1b + S_C1) + 255) & ~(size_t)255;
    auto PBUMP = [&](size_t bytes) -> char* {
        char* p = ws + pcur; pcur = (pcur + bytes + 255) & ~(size_t)255; return p;
    };
    u16b* p32wT = (u16b*)PBUMP(2048UL * 256 * 2);
    u16b* p16wT = (u16b*)PBUMP(512UL * 256 * 2);
    float* pe   = (float*)PBUMP(1024UL * 256 * 4);
    float* ropec= (float*)PBUMP((size_t)NPAD * 32 * 4);
    float* gmean= (float*)PBUMP(64 * 4);
    float* grstd= (float*)PBUMP(64 * 4);
    u16b *wqT[6], *wkT[6], *wvT[6], *pwT[6], *WrT[6], *m1wT[6], *m2wT[6];
    for (int l = 0; l < 6; ++l) {
        wqT[l]  = (u16b*)PBUMP(256UL * 256 * 2);
        wkT[l]  = (u16b*)PBUMP(256UL * 256 * 2);
        wvT[l]  = (u16b*)PBUMP(256UL * 256 * 2);
        pwT[l]  = (u16b*)PBUMP(256UL * 256 * 2);
        WrT[l]  = (u16b*)PBUMP((size_t)NH * RFFD * DKH * 2);
        m1wT[l] = (u16b*)PBUMP(1024UL * 256 * 2);
        m2wT[l] = (u16b*)PBUMP(256UL * 1024 * 2);
    }

    dim3 blk(256);
    auto cdiv = [](long a, long b) -> unsigned { return (unsigned)((a + b - 1) / b); };
    auto gemm = [&](int act, int hasres,
                    const u16b* A, int lda, long sAz, const u16b* Bt, int ldb, long sBz,
                    float* Cf, u16b* Cb, int ldc, long sCz,
                    const float* bias, const float* resid,
                    int tM, int tN, int K, int nz) {
        dim3 g(cdiv((long)tM * tN, 8), 1, nz);
        if (hasres)      gemm_kernel<0,1><<<g, blk, 0, stream>>>(A, lda, sAz, Bt, ldb, sBz, Cf, Cb, ldc, sCz, bias, resid, tM, tN, K);
        else if (act==1) gemm_kernel<1,0><<<g, blk, 0, stream>>>(A, lda, sAz, Bt, ldb, sBz, Cf, Cb, ldc, sCz, bias, resid, tM, tN, K);
        else             gemm_kernel<0,0><<<g, blk, 0, stream>>>(A, lda, sAz, Bt, ldb, sBz, Cf, Cb, ldc, sCz, bias, resid, tM, tN, K);
    };
    auto tpose = [&](const float* src, u16b* dst, int R, int C, long sS, long dS, int nz) {
        dim3 g(cdiv((long)R * C, 256), 1, nz);
        transpose_cvt_kernel<<<g, blk, 0, stream>>>(src, dst, R, C, sS, dS);
    };

    // ---- caches + weight prep (persistent region, safe before stem) ----
    pe2d_kernel<<<1024, blk, 0, stream>>>(pe);
    ropec_kernel<<<cdiv((long)NPAD * 32, 256), blk, 0, stream>>>(ropec);
    tpose(p32w, p32wT, 2048, 256, 0, 0, 1);
    tpose(p16w, p16wT, 512, 256, 0, 0, 1);
    for (int l = 0; l < 6; ++l) {
        tpose(wq[l],  wqT[l],  256, 256, 0, 0, 1);
        tpose(wk[l],  wkT[l],  256, 256, 0, 0, 1);
        tpose(wv[l],  wvT[l],  256, 256, 0, 0, 1);
        tpose(pw[l],  pwT[l],  256, 256, 0, 0, 1);
        tpose(Wr[l],  WrT[l],  DKH, RFFD, (long)DKH * RFFD, (long)RFFD * DKH, NH);
        tpose(m1w[l], m1wT[l], 256, 1024, 0, 0, 1);
        tpose(m2w[l], m2wT[l], 1024, 256, 0, 0, 1);
    }

    // ---- stem ----
    affine_kernel<<<cdiv(8L*4*512*512, 256), blk, 0, stream>>>(grid, chan_s, chan_b, t0);
    // block 0: 512 -> 256
    conv3x3_s2_kernel<<<cdiv(8L*128*256*256, 256), blk, 0, stream>>>(t0, cw[0], c1a, 8, 4, 128, 512, 512);
    gn_stats_kernel<<<64, blk, 0, stream>>>(c1a, gmean, grstd, 128, 256*256);
    gn_gelu_kernel<<<cdiv(8L*128*256*256, 256), blk, 0, stream>>>(c1a, gmean, grstd, g1s[0], g1b[0], 128, 256*256);
    dwconv3x3_kernel<<<cdiv(8L*128*256*256, 256), blk, 0, stream>>>(c1a, dww[0], c1b, 8, 128, 256, 256);
    gn_stats_kernel<<<64, blk, 0, stream>>>(c1b, gmean, grstd, 128, 256*256);
    gn_gelu_kernel<<<cdiv(8L*128*256*256, 256), blk, 0, stream>>>(c1b, gmean, grstd, g2s[0], g2b[0], 128, 256*256);
    // block 1: 256 -> 128  (x32 = c2b)
    conv3x3_s2_kernel<<<cdiv(8L*128*128*128, 256), blk, 0, stream>>>(c1b, cw[1], c2a, 8, 128, 128, 256, 256);
    gn_stats_kernel<<<64, blk, 0, stream>>>(c2a, gmean, grstd, 128, 128*128);
    gn_gelu_kernel<<<cdiv(8L*128*128*128, 256), blk, 0, stream>>>(c2a, gmean, grstd, g1s[1], g1b[1], 128, 128*128);
    dwconv3x3_kernel<<<cdiv(8L*128*128*128, 256), blk, 0, stream>>>(c2a, dww[1], c2b, 8, 128, 128, 128);
    gn_stats_kernel<<<64, blk, 0, stream>>>(c2b, gmean, grstd, 128, 128*128);
    gn_gelu_kernel<<<cdiv(8L*128*128*128, 256), blk, 0, stream>>>(c2b, gmean, grstd, g2s[1], g2b[1], 128, 128*128);
    // block 2: 128 -> 64   (x16 = c3b)
    conv3x3_s2_kernel<<<cdiv(8L*128*64*64, 256), blk, 0, stream>>>(c2b, cw[2], c3a, 8, 128, 128, 128, 128);
    gn_stats_kernel<<<64, blk, 0, stream>>>(c3a, gmean, grstd, 128, 64*64);
    gn_gelu_kernel<<<cdiv(8L*128*64*64, 256), blk, 0, stream>>>(c3a, gmean, grstd, g1s[2], g1b[2], 128, 64*64);
    dwconv3x3_kernel<<<cdiv(8L*128*64*64, 256), blk, 0, stream>>>(c3a, dww[2], c3b, 8, 128, 64, 64);
    gn_stats_kernel<<<64, blk, 0, stream>>>(c3b, gmean, grstd, 128, 64*64);
    gn_gelu_kernel<<<cdiv(8L*128*64*64, 256), blk, 0, stream>>>(c3b, gmean, grstd, g2s[2], g2b[2], 128, 64*64);

    // ---- tokens ----
    patchify_kernel<<<cdiv(8192L*2048, 256), blk, 0, stream>>>(c2b, pm32, 128, 4);
    patchify_kernel<<<cdiv(8192L*512, 256), blk, 0, stream>>>(c3b, pm16, 64, 2);
    zeropad_kernel<<<cdiv(8L*(NPAD-NTOK)*EMB, 256), blk, 0, stream>>>(xseq);
    tcls_kernel<<<8, blk, 0, stream>>>(timesteps, tfw, tfb, clsw, xseq);
    patch_gemm_kernel<<<cdiv(512L*16, 8), blk, 0, stream>>>(pm32, p32wT, p32b, pe, xseq, 2048, 1);
    patch_gemm_kernel<<<cdiv(512L*16, 8), blk, 0, stream>>>(pm16, p16wT, p16b, pe, xseq, 512, 1025);

    // ---- transformer layers ----
    for (int l = 0; l < 6; ++l) {
        layernorm_kernel<0><<<MPAD, blk, 0, stream>>>(xseq, n1s[l], n1b[l], xlnb, nullptr);
        gemm(0, 0, xlnb, EMB, 0, wqT[l], EMB, 0, qf, nullptr, EMB, 0, nullptr, nullptr, TMM, 16, EMB, 1);
        gemm(0, 0, xlnb, EMB, 0, wkT[l], EMB, 0, kf, nullptr, EMB, 0, nullptr, nullptr, TMM, 16, EMB, 1);
        gemm(0, 0, xlnb, EMB, 0, wvT[l], EMB, 0, vf, nullptr, EMB, 0, nullptr, nullptr, TMM, 16, EMB, 1);
        rope_kernel<<<cdiv((long)MPAD*EMB, 256), blk, 0, stream>>>(qf, ropec, qrb);
        rope_kernel<<<cdiv((long)MPAD*EMB, 256), blk, 0, stream>>>(kf, ropec, krb);
        vt_kernel<<<cdiv(8L*NH*DKH*NPAD, 256), blk, 0, stream>>>(vf, Vt);
        {
            dim3 g(cdiv((long)TMM * 8, 8), 1, NH);
            phi_gemm_kernel<0><<<g, blk, 0, stream>>>(qrb, WrT[l], br[l], phiq);
            phi_gemm_kernel<1><<<g, blk, 0, stream>>>(krb, WrT[l], br[l], phiKt);
        }
        ksum_kernel<<<cdiv(8L*NH*F2D, 256), blk, 0, stream>>>(phiKt, ksum);
        // Kv[b,h] = phiKt(F2 x NPAD) @ Vt(DK x NPAD)^T
        gemm(0, 0, phiKt, NPAD, (long)F2D * NPAD, Vt, NPAD, (long)DKH * NPAD,
             Kv, nullptr, DKH, (long)F2D * DKH, nullptr, nullptr, F2D/16, DKH/16, NPAD, 64);
        tpose(Kv, Kvt, F2D, DKH, (long)F2D * DKH, (long)F2D * DKH, 64);
        denom_kernel<<<cdiv(8L*NPAD*NH, 256), blk, 0, stream>>>(phiq, ksum, denom);
        {
            dim3 g(cdiv((long)(NPAD/16) * 2, 8), 1, 64);
            attn_gemm_kernel<<<g, blk, 0, stream>>>(phiq, Kvt, denom, attnb);
        }
        gemm(0, 1, attnb, EMB, 0, pwT[l], EMB, 0, xseq, nullptr, EMB, 0, pb[l], xseq, TMM, 16, EMB, 1);
        layernorm_kernel<0><<<MPAD, blk, 0, stream>>>(xseq, n2s[l], n2b[l], xlnb, nullptr);
        gemm(1, 0, xlnb, EMB, 0, m1wT[l], EMB, 0, nullptr, h1, 1024, 0, m1b[l], nullptr, TMM, 64, EMB, 1);
        gemm(0, 1, h1, 1024, 0, m2wT[l], 1024, 0, xseq, nullptr, EMB, 0, m2b[l], xseq, TMM, 16, 1024, 1);
    }

    // ---- final norm -> output (cls | tokens) ----
    layernorm_kernel<1><<<MPAD, blk, 0, stream>>>(xseq, post_s, post_b, nullptr, (float*)d_out);
}
